// LRFBlock_72232759984879
// MI455X (gfx1250) — compile-verified, matrix operand order checked by reading, not run on hardware
//
#include <hip/hip_runtime.h>
#include <hip/hip_bf16.h>
#include <cstdint>

// ---------------------------------------------------------------------------
// Types for CDNA5 WMMA (gfx1250, wave32)
// ---------------------------------------------------------------------------
typedef __attribute__((ext_vector_type(16))) __bf16 v16bf;
typedef __attribute__((ext_vector_type(8)))  float  v8f;

#define ALPHA 1.0f

__device__ __forceinline__ __bf16 f2bf(float f) {
    union { float f; unsigned u; } a; a.f = f;
    unsigned r = a.u + 0x7FFFu + ((a.u >> 16) & 1u);   // round-to-nearest-even
    unsigned short h = (unsigned short)(r >> 16);
    __bf16 o; __builtin_memcpy(&o, &h, 2); return o;
}
__device__ __forceinline__ float bf2f(__bf16 b) {
    unsigned short h; __builtin_memcpy(&h, &b, 2);
    union { unsigned u; float f; } a; a.u = ((unsigned)h) << 16;
    return a.f;
}

__device__ __forceinline__ v8f wmma_bf16(v16bf a, v16bf b, v8f c) {
    // D = A(16x32 bf16) * B(32x16 bf16) + C(16x16 f32)
    return __builtin_amdgcn_wmma_f32_16x16x32_bf16(false, a, false, b,
                                                   (short)0, c, false, false);
}

union FragU { uint4 u[2]; v16bf v; };

// Load a 16-lane-striped bf16 fragment (ISA 7.12.2 16-bit A layout).
// p must already point at (row_base + lane-specific k offset); ld in elements.
__device__ __forceinline__ v16bf load_frag(const __bf16* p) {
    FragU t;
    t.u[0] = *(const uint4*)(p);        // K = kb .. kb+7
    t.u[1] = *(const uint4*)(p + 16);   // K = kb+16 .. kb+23
    return t.v;
}

// ---------------------------------------------------------------------------
// Elementwise fp32 -> bf16 convert
// ---------------------------------------------------------------------------
__global__ __launch_bounds__(256) void cvt_kernel(const float* __restrict__ in,
                                                  __bf16* __restrict__ out,
                                                  long long n) {
    long long i = (long long)blockIdx.x * 256 + threadIdx.x;
    if (i < n) out[i] = f2bf(in[i]);
}

// ---------------------------------------------------------------------------
// Transpose + convert: in [K,N] fp32 -> out [N,K] bf16 (K,N multiples of 32)
// ---------------------------------------------------------------------------
__global__ __launch_bounds__(256) void transpose_cvt_kernel(
    const float* __restrict__ in, __bf16* __restrict__ out, int K, int N) {
    __shared__ float tile[32][33];
    int tx = threadIdx.x & 31, ty = threadIdx.x >> 5;
    int n0 = blockIdx.x * 32, k0 = blockIdx.y * 32;
    #pragma unroll
    for (int r = ty; r < 32; r += 8)
        tile[r][tx] = in[(size_t)(k0 + r) * N + (n0 + tx)];
    __syncthreads();
    #pragma unroll
    for (int r = ty; r < 32; r += 8)
        out[(size_t)(n0 + r) * K + (k0 + tx)] = f2bf(tile[tx][r]);
}

// ---------------------------------------------------------------------------
// Tiled bf16 WMMA GEMM:  C[M,N] = A[M,K] @ Bt[N,K]^T + bias  (+epilogue)
// Block = 256 threads = 8 waves, tile 128x128, BK=32. Wave tile 32x64.
// EPI: 0 = bias -> bf16 out ; 1 = bias + residual -> f32 out ;
//      2 = bias + exact GELU -> bf16 out
// ---------------------------------------------------------------------------
constexpr int BM = 128, BN = 128, BKT = 32, LDT = 40;

template <int EPI>
__global__ __launch_bounds__(256) void gemm_bf16_kernel(
    const __bf16* __restrict__ A, const __bf16* __restrict__ Bt,
    const float* __restrict__ bias, const float* __restrict__ res,
    float* __restrict__ outF, __bf16* __restrict__ outB,
    int M, int N, int K) {
    __shared__ __bf16 As[BM * LDT];
    __shared__ __bf16 Bs[BN * LDT];

    const int tid  = threadIdx.x;
    const int lane = tid & 31, w = tid >> 5;
    const int wm = w & 3, wn = w >> 2;            // 4 x 2 wave grid
    const int m0 = blockIdx.y * BM, n0 = blockIdx.x * BN;
    const int l15 = lane & 15;
    const int kb  = (lane >> 4) * 8;              // per-lane K offset
    const int hi8 = (lane >> 4) * 8;              // C-layout row offset

    const int srow = tid >> 1, scol = (tid & 1) * 16;   // staging map

    v8f acc[2][4];
    #pragma unroll
    for (int s = 0; s < 2; ++s)
        #pragma unroll
        for (int n = 0; n < 4; ++n)
            #pragma unroll
            for (int i = 0; i < 8; ++i) acc[s][n][i] = 0.0f;

    for (int k0 = 0; k0 < K; k0 += BKT) {
        const __bf16* ga = A  + (size_t)(m0 + srow) * K + k0 + scol;
        const __bf16* gb = Bt + (size_t)(n0 + srow) * K + k0 + scol;
        *(uint4*)&As[srow * LDT + scol]     = *(const uint4*)(ga);
        *(uint4*)&As[srow * LDT + scol + 8] = *(const uint4*)(ga + 8);
        *(uint4*)&Bs[srow * LDT + scol]     = *(const uint4*)(gb);
        *(uint4*)&Bs[srow * LDT + scol + 8] = *(const uint4*)(gb + 8);
        __builtin_prefetch(ga + BKT, 0, 1);        // global_prefetch_b8
        __builtin_prefetch(gb + BKT, 0, 1);
        __syncthreads();

        v16bf af[2], bf[4];
        #pragma unroll
        for (int s = 0; s < 2; ++s)
            af[s] = load_frag(&As[(wm * 32 + s * 16 + l15) * LDT + kb]);
        #pragma unroll
        for (int n = 0; n < 4; ++n)
            bf[n] = load_frag(&Bs[(wn * 64 + n * 16 + l15) * LDT + kb]);
        #pragma unroll
        for (int s = 0; s < 2; ++s)
            #pragma unroll
            for (int n = 0; n < 4; ++n)
                acc[s][n] = wmma_bf16(af[s], bf[n], acc[s][n]);
        __syncthreads();
    }

    #pragma unroll
    for (int s = 0; s < 2; ++s)
        #pragma unroll
        for (int n = 0; n < 4; ++n) {
            const int gc = n0 + wn * 64 + n * 16 + l15;
            const float bv = bias[gc];
            #pragma unroll
            for (int i = 0; i < 8; ++i) {
                const int gr = m0 + wm * 32 + s * 16 + i + hi8;
                const size_t off = (size_t)gr * N + gc;
                float v = acc[s][n][i] + bv;
                if (EPI == 0) {
                    outB[off] = f2bf(v);
                } else if (EPI == 1) {
                    outF[off] = v + res[off];
                } else {
                    float g = 0.5f * v * (1.0f + erff(v * 0.70710678118654752f));
                    outB[off] = f2bf(g);
                }
            }
        }
}

// ---------------------------------------------------------------------------
// Per-(b,h) attention stats from bf16 qkv [B,N,3D]:
//   stats[(b*H+h)*3+0] = (max_n sum_d |q|)^2
//   stats[(b*H+h)*3+1] = (max_d sum_n |k|)^2
//   stats[(b*H+h)*3+2] = ||q||_F
// ---------------------------------------------------------------------------
__global__ __launch_bounds__(256) void attn_stats_kernel(
    const __bf16* __restrict__ qkv, float* __restrict__ stats, int Nn) {
    const int D = 768, TD = 2304, H = 12, HD = 64;
    const int b = blockIdx.y, h = blockIdx.x, tid = threadIdx.x;
    __shared__ float redM[256];
    __shared__ float redQ[256];
    __shared__ float colred[64];

    float maxrow = 0.f, sumsq = 0.f;
    for (int n = tid; n < Nn; n += 256) {
        const __bf16* qp = qkv + ((size_t)(b * Nn + n)) * TD + h * HD;
        float rs = 0.f;
        #pragma unroll 8
        for (int d = 0; d < HD; ++d) {
            float v = bf2f(qp[d]);
            rs += fabsf(v);
            sumsq += v * v;
        }
        maxrow = fmaxf(maxrow, rs);
    }
    redM[tid] = maxrow; redQ[tid] = sumsq;
    __syncthreads();
    for (int s = 128; s > 0; s >>= 1) {
        if (tid < s) {
            redM[tid] = fmaxf(redM[tid], redM[tid + s]);
            redQ[tid] += redQ[tid + s];
        }
        __syncthreads();
    }
    if (tid < HD) {
        float cs = 0.f;
        for (int n = 0; n < Nn; ++n)
            cs += fabsf(bf2f(qkv[((size_t)(b * Nn + n)) * TD + D + h * HD + tid]));
        colred[tid] = cs;
    }
    __syncthreads();
    if (tid == 0) {
        float cm = 0.f;
        for (int i = 0; i < 64; ++i) cm = fmaxf(cm, colred[i]);
        float mr = redM[0];
        stats[(b * H + h) * 3 + 0] = mr * mr;
        stats[(b * H + h) * 3 + 1] = cm * cm;
        stats[(b * H + h) * 3 + 2] = sqrtf(redQ[0]);
    }
}

// x_inf[b] = max_d sum_n |x[b,n,d]| , x fp32 [B,N,D]
__global__ __launch_bounds__(256) void xinf_kernel(const float* __restrict__ x,
                                                   float* __restrict__ xinf,
                                                   int Nn) {
    const int D = 768;
    const int b = blockIdx.x, tid = threadIdx.x;
    __shared__ float red[256];
    float best = 0.f;
    for (int d = tid; d < D; d += 256) {
        float cs = 0.f;
        for (int n = 0; n < Nn; ++n)
            cs += fabsf(x[((size_t)(b * Nn + n)) * D + d]);
        best = fmaxf(best, cs);
    }
    red[tid] = best;
    __syncthreads();
    for (int s = 128; s > 0; s >>= 1) {
        if (tid < s) red[tid] = fmaxf(red[tid], red[tid + s]);
        __syncthreads();
    }
    if (tid == 0) xinf[b] = red[0];
}

// ---------------------------------------------------------------------------
// Flash-style LRS attention. qkv bf16 [B,N,3D]; out bf16 [B,N,D].
// Grid (N/128, H, B); 256 threads = 8 waves; each wave owns 16 query rows.
// sim = scale*qk + cconst (q_row_2/k_col_2 are per-(b,h) scalars).
// ---------------------------------------------------------------------------
__global__ __launch_bounds__(256) void attn_kernel(
    const __bf16* __restrict__ qkv, const float* __restrict__ stats,
    const float* __restrict__ xinf, __bf16* __restrict__ outw, int Nn) {
    const int D = 768, TD = 2304, H = 12, HD = 64;
    const int b = blockIdx.z, h = blockIdx.y;
    const int tid = threadIdx.x, w = tid >> 5, lane = tid & 31;
    const int row0 = blockIdx.x * 128 + w * 16;
    const int l15 = lane & 15;
    const int kb  = (lane >> 4) * 8;
    const int hi8 = (lane >> 4) * 8;

    __shared__ __bf16 Kt[32 * 72];       // [key][dim]   (B-frag source for QK)
    __shared__ __bf16 Vt[64 * 40];       // [dim][key]   (B-frag source for PV)
    __shared__ __bf16 Pl[8][16 * 40];    // per-wave P bounce (C-layout -> A-frag)

    const float q2 = stats[(b * H + h) * 3 + 0];
    const float k2 = stats[(b * H + h) * 3 + 1];
    const float qf = stats[(b * H + h) * 3 + 2];
    const float denom = qf * xinf[b] + 1e-20f;
    const float scale = 2.0f * ALPHA / denom;
    const float cc    = -ALPHA * (q2 + k2) / denom;

    // Q fragments (16 rows x 64 dims = two 16x32 A-frags), direct from global
    v16bf qa[2];
    {
        const int qrow = row0 + l15;
        const __bf16* qp = qkv + ((size_t)(b * Nn + qrow)) * TD + h * HD;
        #pragma unroll
        for (int c = 0; c < 2; ++c) {
            FragU t;
            t.u[0] = *(const uint4*)(qp + c * 32 + kb);
            t.u[1] = *(const uint4*)(qp + c * 32 + kb + 16);
            qa[c] = t.v;
        }
    }

    float mrow[8], srow[8];
    v8f o[4];
    #pragma unroll
    for (int i = 0; i < 8; ++i) { mrow[i] = -1e30f; srow[i] = 0.f; }
    #pragma unroll
    for (int n = 0; n < 4; ++n)
        #pragma unroll
        for (int i = 0; i < 8; ++i) o[n][i] = 0.f;

    for (int kt = 0; kt < Nn; kt += 32) {
        // --- stage K tile straight, V tile transposed ---
        {
            const int key = tid >> 3, dd = (tid & 7) * 8;
            const __bf16* kp = qkv + ((size_t)(b * Nn + kt + key)) * TD + D + h * HD + dd;
            *(uint4*)&Kt[key * 72 + dd] = *(const uint4*)kp;
            const __bf16* vp = qkv + ((size_t)(b * Nn + kt + key)) * TD + 2 * D + h * HD + dd;
            uint4 vv = *(const uint4*)vp;
            const __bf16* ve = (const __bf16*)&vv;
            #pragma unroll
            for (int e = 0; e < 8; ++e) Vt[(dd + e) * 40 + key] = ve[e];
        }
        __syncthreads();

        // --- S = Q K^T over 32 keys (two 16x16 tiles) ---
        v8f s[2];
        #pragma unroll
        for (int j = 0; j < 2; ++j) {
            #pragma unroll
            for (int i = 0; i < 8; ++i) s[j][i] = 0.f;
            #pragma unroll
            for (int c = 0; c < 2; ++c) {
                v16bf kf = load_frag(&Kt[(j * 16 + l15) * 72 + c * 32 + kb]);
                s[j] = wmma_bf16(qa[c], kf, s[j]);
            }
        }

        // --- online softmax update ---
        float corr[8];
        #pragma unroll
        for (int i = 0; i < 8; ++i) {
            float a0 = s[0][i] * scale + cc;
            float a1 = s[1][i] * scale + cc;
            float m = fmaxf(a0, a1);
            #pragma unroll
            for (int d = 1; d < 16; d <<= 1) m = fmaxf(m, __shfl_xor(m, d, 32));
            float mnew = fmaxf(mrow[i], m);
            corr[i] = __expf(mrow[i] - mnew);
            float p0 = __expf(a0 - mnew);
            float p1 = __expf(a1 - mnew);
            s[0][i] = p0; s[1][i] = p1;
            float pp = p0 + p1;
            #pragma unroll
            for (int d = 1; d < 16; d <<= 1) pp += __shfl_xor(pp, d, 32);
            srow[i] = srow[i] * corr[i] + pp;
            mrow[i] = mnew;
        }
        #pragma unroll
        for (int n = 0; n < 4; ++n)
            #pragma unroll
            for (int i = 0; i < 8; ++i) o[n][i] *= corr[i];

        // --- P: C-layout -> A-frag layout via per-wave LDS bounce ---
        #pragma unroll
        for (int j = 0; j < 2; ++j)
            #pragma unroll
            for (int i = 0; i < 8; ++i)
                Pl[w][(i + hi8) * 40 + j * 16 + l15] = f2bf(s[j][i]);
        __syncthreads();
        v16bf pf = load_frag(&Pl[w][l15 * 40 + kb]);

        // --- O += P @ V ---
        #pragma unroll
        for (int n = 0; n < 4; ++n) {
            v16bf vf = load_frag(&Vt[(n * 16 + l15) * 40 + kb]);
            o[n] = wmma_bf16(pf, vf, o[n]);
        }
        __syncthreads();
    }

    // normalize + store bf16
    #pragma unroll
    for (int n = 0; n < 4; ++n)
        #pragma unroll
        for (int i = 0; i < 8; ++i) {
            const int r = row0 + i + hi8;
            const int c = h * HD + n * 16 + l15;
            outw[((size_t)(b * Nn + r)) * D + c] = f2bf(o[n][i] / srow[i]);
        }
}

// ---------------------------------------------------------------------------
// LayerNorm over D=768 (one block/row, 256 threads x 3 elems)
// ---------------------------------------------------------------------------
__global__ __launch_bounds__(256) void ln_kernel(
    const float* __restrict__ in, const float* __restrict__ g,
    const float* __restrict__ be, float* __restrict__ outF,
    __bf16* __restrict__ outB) {
    const int D = 768;
    const size_t row = blockIdx.x;
    const int tid = threadIdx.x;
    const float* p = in + row * D;
    __shared__ float rs[256];
    __shared__ float rq[256];
    float vals[3], s = 0.f, q = 0.f;
    #pragma unroll
    for (int j = 0; j < 3; ++j) {
        float v = p[tid + j * 256];
        vals[j] = v; s += v; q += v * v;
    }
    rs[tid] = s; rq[tid] = q;
    __syncthreads();
    for (int st = 128; st > 0; st >>= 1) {
        if (tid < st) { rs[tid] += rs[tid + st]; rq[tid] += rq[tid + st]; }
        __syncthreads();
    }
    const float mu  = rs[0] * (1.0f / D);
    const float var = rq[0] * (1.0f / D) - mu * mu;
    const float inv = rsqrtf(var + 1e-5f);
    #pragma unroll
    for (int j = 0; j < 3; ++j) {
        const int c = tid + j * 256;
        float o = (vals[j] - mu) * inv * g[c] + be[c];
        if (outF) outF[row * D + c] = o;
        if (outB) outB[row * D + c] = f2bf(o);
    }
}

// ---------------------------------------------------------------------------
// Launcher
// ---------------------------------------------------------------------------
extern "C" void kernel_launch(void* const* d_in, const int* in_sizes, int n_in,
                              void* d_out, int out_size, void* d_ws, size_t ws_size,
                              hipStream_t stream) {
    (void)in_sizes; (void)n_in; (void)out_size; (void)ws_size;
    const int   Bn = 8, Nn = 1024, Dd = 768, Hh = 12, TDd = 2304, HIDd = 3072;
    const int   M = Bn * Nn;                       // 8192 rows

    const float* x       = (const float*)d_in[0];
    const float* qkv_w1  = (const float*)d_in[1];
    const float* qkv_b1  = (const float*)d_in[2];
    const float* proj_w1 = (const float*)d_in[3];
    const float* proj_b1 = (const float*)d_in[4];
    const float* qkv_w2  = (const float*)d_in[5];
    const float* qkv_b2  = (const float*)d_in[6];
    const float* proj_w2 = (const float*)d_in[7];
    const float* proj_b2 = (const float*)d_in[8];
    const float* n1g = (const float*)d_in[9],  *n1b = (const float*)d_in[10];
    const float* n2g = (const float*)d_in[11], *n2b = (const float*)d_in[12];
    const float* n3g = (const float*)d_in[13], *n3b = (const float*)d_in[14];
    const float* fc1_w = (const float*)d_in[15], *fc1_b = (const float*)d_in[16];
    const float* fc2_w = (const float*)d_in[17], *fc2_b = (const float*)d_in[18];

    // workspace carve-up
    char* ws = (char*)d_ws;
    size_t off = 0;
    auto take = [&](size_t bytes) -> char* {
        char* p = ws + off;
        off = (off + bytes + 255) & ~(size_t)255;
        return p;
    };
    __bf16* xbf     = (__bf16*)take((size_t)M * Dd * 2);
    __bf16* wqkv1t  = (__bf16*)take((size_t)TDd * Dd * 2);
    __bf16* wqkv2t  = (__bf16*)take((size_t)TDd * Dd * 2);
    __bf16* wproj1t = (__bf16*)take((size_t)Dd * Dd * 2);
    __bf16* wproj2t = (__bf16*)take((size_t)Dd * Dd * 2);
    __bf16* wfc1t   = (__bf16*)take((size_t)HIDd * Dd * 2);
    __bf16* wfc2t   = (__bf16*)take((size_t)Dd * HIDd * 2);
    __bf16* qkv1    = (__bf16*)take((size_t)M * TDd * 2);
    __bf16* qkv2    = (__bf16*)take((size_t)M * TDd * 2);
    __bf16* attw1   = (__bf16*)take((size_t)M * Dd * 2);
    __bf16* attw2   = (__bf16*)take((size_t)M * Dd * 2);
    float*  y1      = (float*)take((size_t)M * Dd * 4);
    float*  y2      = (float*)take((size_t)M * Dd * 4);
    float*  x1f     = (float*)take((size_t)M * Dd * 4);
    __bf16* x2bf    = (__bf16*)take((size_t)M * Dd * 2);
    __bf16* hbf     = (__bf16*)take((size_t)M * HIDd * 2);
    float*  y3      = (float*)take((size_t)M * Dd * 4);
    float*  stats1  = (float*)take((size_t)Bn * Hh * 3 * 4);
    float*  stats2  = (float*)take((size_t)Bn * Hh * 3 * 4);
    float*  xinf    = (float*)take((size_t)Bn * 4);

    // 1) conversions
    {
        long long n = (long long)M * Dd;
        cvt_kernel<<<(unsigned)((n + 255) / 256), 256, 0, stream>>>(x, xbf, n);
    }
    transpose_cvt_kernel<<<dim3(TDd / 32, Dd / 32), 256, 0, stream>>>(qkv_w1, wqkv1t, Dd, TDd);
    transpose_cvt_kernel<<<dim3(TDd / 32, Dd / 32), 256, 0, stream>>>(qkv_w2, wqkv2t, Dd, TDd);
    transpose_cvt_kernel<<<dim3(Dd / 32, Dd / 32), 256, 0, stream>>>(proj_w1, wproj1t, Dd, Dd);
    transpose_cvt_kernel<<<dim3(Dd / 32, Dd / 32), 256, 0, stream>>>(proj_w2, wproj2t, Dd, Dd);
    transpose_cvt_kernel<<<dim3(HIDd / 32, Dd / 32), 256, 0, stream>>>(fc1_w, wfc1t, Dd, HIDd);
    transpose_cvt_kernel<<<dim3(Dd / 32, HIDd / 32), 256, 0, stream>>>(fc2_w, wfc2t, HIDd, Dd);

    // 2) QKV GEMMs (bf16 out)
    gemm_bf16_kernel<0><<<dim3(TDd / BN, M / BM), 256, 0, stream>>>(
        xbf, wqkv1t, qkv_b1, nullptr, nullptr, qkv1, M, TDd, Dd);
    gemm_bf16_kernel<0><<<dim3(TDd / BN, M / BM), 256, 0, stream>>>(
        xbf, wqkv2t, qkv_b2, nullptr, nullptr, qkv2, M, TDd, Dd);

    // 3) norms / stats
    attn_stats_kernel<<<dim3(Hh, Bn), 256, 0, stream>>>(qkv1, stats1, Nn);
    attn_stats_kernel<<<dim3(Hh, Bn), 256, 0, stream>>>(qkv2, stats2, Nn);
    xinf_kernel<<<Bn, 256, 0, stream>>>(x, xinf, Nn);

    // 4) attention (flash-style WMMA)
    attn_kernel<<<dim3(Nn / 128, Hh, Bn), 256, 0, stream>>>(qkv1, stats1, xinf, attw1, Nn);
    attn_kernel<<<dim3(Nn / 128, Hh, Bn), 256, 0, stream>>>(qkv2, stats2, xinf, attw2, Nn);

    // 5) projections + residual (f32 out)
    gemm_bf16_kernel<1><<<dim3(Dd / BN, M / BM), 256, 0, stream>>>(
        attw1, wproj1t, proj_b1, x, y1, nullptr, M, Dd, Dd);
    gemm_bf16_kernel<1><<<dim3(Dd / BN, M / BM), 256, 0, stream>>>(
        attw2, wproj2t, proj_b2, x, y2, nullptr, M, Dd, Dd);

    // 6) layernorms
    ln_kernel<<<M, 256, 0, stream>>>(y1, n1g, n1b, x1f, nullptr);
    ln_kernel<<<M, 256, 0, stream>>>(y2, n2g, n2b, nullptr, x2bf);

    // 7) MLP
    gemm_bf16_kernel<2><<<dim3(HIDd / BN, M / BM), 256, 0, stream>>>(
        x2bf, wfc1t, fc1_b, nullptr, nullptr, hbf, M, HIDd, Dd);
    gemm_bf16_kernel<1><<<dim3(Dd / BN, M / BM), 256, 0, stream>>>(
        hbf, wfc2t, fc2_b, x1f, y3, nullptr, M, Dd, HIDd);

    // 8) final layernorm -> output (fp32)
    ln_kernel<<<M, 256, 0, stream>>>(y3, n3g, n3b, (float*)d_out, nullptr);
}
